// RNNLM_48335561949299
// MI455X (gfx1250) — matrix operand order repaired and checked by reading, not compiled, not run on hardware
//
#include <hip/hip_runtime.h>
#include <hip/hip_bf16.h>
#include <math.h>

#define VOCAB 32000
#define EMB   512
#define HID   1024
#define BB    16
#define TT    512

typedef float  v8f  __attribute__((ext_vector_type(8)));
typedef float  v2f  __attribute__((ext_vector_type(2)));
typedef __bf16 bf16;
typedef bf16   v16bf __attribute__((ext_vector_type(16)));
typedef bf16   v8bf  __attribute__((ext_vector_type(8)));

#if defined(__has_builtin)
#if __has_builtin(__builtin_amdgcn_global_load_async_to_lds_b128)
#define HAVE_ASYNC_LDS 1
#endif
#endif

// exact pointee type the async-LDS builtin expects (per compiler diagnostic):
// non-const GCC-vector int4, in AS1 (global) / AS3 (LDS)
typedef int gv4i __attribute__((vector_size(16)));
typedef __attribute__((address_space(1))) gv4i as1_v4i;
typedef __attribute__((address_space(3))) gv4i as3_v4i;

__device__ __forceinline__ void wait_async0() {
#if defined(__has_builtin) && __has_builtin(__builtin_amdgcn_s_wait_asynccnt)
  __builtin_amdgcn_s_wait_asynccnt(0);
#else
  asm volatile("s_wait_asynccnt 0" ::: "memory");
#endif
}

// ---------------------------------------------------------------------------
// Kernel 0: lin_w f32 -> bf16 (lin_w is L2-resident; one-time convert keeps
// the 537-GFLOP output GEMM on the 16x16x32 bf16 WMMA pipe). Also resets the
// global-barrier state each launch (deterministic across graph replays).
// ---------------------------------------------------------------------------
__global__ void k_convert_w(const float* __restrict__ lw, bf16* __restrict__ lwb,
                            int n, int* __restrict__ sync) {
  size_t i = (size_t)blockIdx.x * blockDim.x + threadIdx.x;
  size_t stride = (size_t)gridDim.x * blockDim.x;
  for (; i < (size_t)n; i += stride) lwb[i] = (bf16)lw[i];
  if (blockIdx.x == 0 && threadIdx.x == 0) { sync[0] = 0; sync[1] = 0; }
}

// ---------------------------------------------------------------------------
// Kernel 1: xW[t][b][h] = sum_e embedding[X[b][t]][e] * W_x[h][e]
// f32 WMMA 16x16x4 (full precision feeds the chaotic recurrence).
// One wave = one 16(batch) x 64(hidden) strip (4 accumulators reuse A frag).
// ---------------------------------------------------------------------------
__global__ void k_embed_xw(const int* __restrict__ X, const float* __restrict__ emb,
                           const float* __restrict__ Wx, float* __restrict__ xW) {
  const int lane  = threadIdx.x & 31;
  const int wave  = threadIdx.x >> 5;
  const int wg    = blockIdx.x * 8 + wave;    // 0..8191
  const int t     = wg >> 4;                  // 0..511
  const int hbase = (wg & 15) * 64;           // 0..960
  const int l15   = lane & 15;
  const int half  = lane >> 4;

  const int row = X[l15 * TT + t];            // X is [B,T]
  const float* arow = emb + (size_t)row * EMB;
  const float* brow[4];
#pragma unroll
  for (int j = 0; j < 4; ++j) brow[j] = Wx + (size_t)(hbase + j * 16 + l15) * EMB;

  v8f acc[4];
#pragma unroll
  for (int j = 0; j < 4; ++j) acc[j] = (v8f){0.f,0.f,0.f,0.f,0.f,0.f,0.f,0.f};

  for (int kb = 0; kb < EMB; kb += 4) {
    const int ko = kb + 2 * half;
    v2f a = *(const v2f*)(arow + ko);
#pragma unroll
    for (int j = 0; j < 4; ++j) {
      v2f w = *(const v2f*)(brow[j] + ko);
      acc[j] = __builtin_amdgcn_wmma_f32_16x16x4_f32(false, a, false, w,
                                                     (short)0, acc[j], false, false);
    }
  }
#pragma unroll
  for (int j = 0; j < 4; ++j) {
    const int n = hbase + j * 16 + l15;
#pragma unroll
    for (int r = 0; r < 8; ++r) {
      const int m = r + 8 * half;             // batch index
      xW[((size_t)t * BB + m) * HID + n] = acc[j][r];
    }
  }
}

// ---------------------------------------------------------------------------
// Global barrier for the persistent recurrence kernel (8 WGs -> cheap).
// ---------------------------------------------------------------------------
__device__ __forceinline__ void grid_barrier(int* cnt, int* gen, int nwg) {
  __threadfence();
  __syncthreads();
  if (threadIdx.x == 0) {
    const int my = __hip_atomic_load(gen, __ATOMIC_RELAXED, __HIP_MEMORY_SCOPE_AGENT);
    const int arrived =
        __hip_atomic_fetch_add(cnt, 1, __ATOMIC_ACQ_REL, __HIP_MEMORY_SCOPE_AGENT) + 1;
    if (arrived == nwg) {
      __hip_atomic_store(cnt, 0, __ATOMIC_RELAXED, __HIP_MEMORY_SCOPE_AGENT);
      __hip_atomic_fetch_add(gen, 1, __ATOMIC_RELEASE, __HIP_MEMORY_SCOPE_AGENT);
    } else {
      while (__hip_atomic_load(gen, __ATOMIC_ACQUIRE, __HIP_MEMORY_SCOPE_AGENT) == my)
        __builtin_amdgcn_s_sleep(1);
    }
  }
  __syncthreads();
}

// ---------------------------------------------------------------------------
// Kernel 2: persistent RNN scan, 8 WGs x 8 waves. Wave w of WG g owns the
// 16x16 tile of h covering columns [(g*8+w)*16, +16), full K=1024 per step
// (no cross-wave reduction). f32 WMMA, 4 interleaved accumulators to break
// the WMMA->WMMA RAW chain. Emits h (f32 ping-pong) and hs (bf16 for GEMM).
// ---------------------------------------------------------------------------
__global__ void k_rnn(const float* __restrict__ xW, const float* __restrict__ Wh,
                      float* __restrict__ hbuf, bf16* __restrict__ hsb,
                      int* __restrict__ sync) {
  int* cnt = sync;
  int* gen = sync + 1;
  const int lane = threadIdx.x & 31;
  const int wave = threadIdx.x >> 5;
  const int tile = blockIdx.x * 8 + wave;     // 0..63
  const int l15  = lane & 15;
  const int half = lane >> 4;
  const int n    = tile * 16 + l15;           // this lane's output column
  const float* wrow = Wh + (size_t)n * HID;   // B[k][n] = W_h[n][k]

  // zero h0 (this WG's 128 columns), then sync before step 0
  {
    const int c0 = blockIdx.x * 128;
    for (int i = threadIdx.x; i < BB * 128; i += blockDim.x)
      hbuf[(size_t)(i >> 7) * HID + c0 + (i & 127)] = 0.f;
  }
  grid_barrier(cnt, gen, gridDim.x);

  for (int t = 0; t < TT; ++t) {
    const float* hc = hbuf + (size_t)(t & 1) * BB * HID;
    float*       hn = hbuf + (size_t)((t & 1) ^ 1) * BB * HID;
    const float* hrow = hc + (size_t)l15 * HID;   // A row m = l15

    v8f acc[4];
#pragma unroll
    for (int c = 0; c < 4; ++c) acc[c] = (v8f){0.f,0.f,0.f,0.f,0.f,0.f,0.f,0.f};

#pragma unroll 4
    for (int kb = 0; kb < HID; kb += 16) {
#pragma unroll
      for (int c = 0; c < 4; ++c) {
        const int ko = kb + 4 * c + 2 * half;
        v2f a = *(const v2f*)(hrow + ko);
        v2f w = *(const v2f*)(wrow + ko);
        acc[c] = __builtin_amdgcn_wmma_f32_16x16x4_f32(false, a, false, w,
                                                       (short)0, acc[c], false, false);
      }
    }
    v8f s = (acc[0] + acc[1]) + (acc[2] + acc[3]);

#pragma unroll
    for (int r = 0; r < 8; ++r) {
      const int m = r + 8 * half;               // batch index
      float v = s[r] + xW[((size_t)t * BB + m) * HID + n];
      v = tanhf(v);
      hn[(size_t)m * HID + n] = v;
      hsb[((size_t)m * TT + t) * HID + n] = (bf16)v;  // hs[b][t][h]
    }
    grid_barrier(cnt, gen, gridDim.x);
  }
}

// ---------------------------------------------------------------------------
// Kernel 3: logits = hs(bf16) @ lin_w(bf16)^T + b, f32 accumulate.
// bf16 WMMA 16x16x32. Block = 8 waves stacked in M -> 128(M) x 128(N) tile.
// The shared 128-col lin_w panel is staged per 64-k chunk into LDS via the
// CDNA5 async Global->LDS path (ASYNCcnt) so the 8 waves hit LDS instead of
// producing 8x redundant L2 traffic. Column stride padded to 160B to spread
// LDS banks for the 32B v16bf fragment reads.
// ---------------------------------------------------------------------------
#define KC      64          // k-chunk staged per iteration
#define BSTRIDE 80          // padded LDS column stride in elements (160 B)

__global__ void k_logits(const bf16* __restrict__ hsb, const bf16* __restrict__ lwb,
                         const float* __restrict__ lb, float* __restrict__ out) {
  __shared__ bf16 Bs[128 * BSTRIDE];          // 20 KB

  const int lane  = threadIdx.x & 31;
  const int wave  = threadIdx.x >> 5;
  const int l15   = lane & 15;
  const int half  = lane >> 4;
  const int mtile = blockIdx.y * 8 + wave;    // 0..511
  const int nbase = blockIdx.x * 128;         // 0..31872
  const int m0    = mtile * 16;

  // staging assignment: thread -> (column, 16B part), 4 passes of 32 columns
  const int scol  = threadIdx.x >> 3;         // 0..31
  const int spart = threadIdx.x & 7;          // 0..7 -> k offset spart*8

  const bf16* arow = hsb + (size_t)(m0 + l15) * HID;  // A row m = l15
  v8f acc[8];
#pragma unroll
  for (int j = 0; j < 8; ++j) acc[j] = (v8f){0.f,0.f,0.f,0.f,0.f,0.f,0.f,0.f};

  for (int kb = 0; kb < HID; kb += KC) {
    // ---- stage B chunk: 128 cols x KC k (lin_w rows nbase..nbase+127) ----
#pragma unroll
    for (int p = 0; p < 4; ++p) {
      const int col = scol + p * 32;
      const bf16* gsrc = lwb + (size_t)(nbase + col) * HID + kb + spart * 8;
      bf16* ldst = Bs + col * BSTRIDE + spart * 8;
#ifdef HAVE_ASYNC_LDS
      __builtin_amdgcn_global_load_async_to_lds_b128(
          (as1_v4i*)(uintptr_t)gsrc,
          (as3_v4i*)(unsigned)(uintptr_t)ldst, 0, 0);
#else
      *(uint4*)ldst = *(const uint4*)gsrc;    // load + ds_store_b128 fallback
#endif
    }
#ifdef HAVE_ASYNC_LDS
    wait_async0();
#endif
    __syncthreads();

    // ---- consume: 2 k-steps of 32, 8 N-tiles each ----
#pragma unroll
    for (int kc = 0; kc < KC; kc += 32) {
      // A frag: k = kc + 16*half .. +15 contiguous (matches B)
      v8bf a0 = *(const v8bf*)(arow + kb + kc + 16 * half);
      v8bf a1 = *(const v8bf*)(arow + kb + kc + 16 * half + 8);
      v16bf a = __builtin_shufflevector(a0, a1, 0,1,2,3,4,5,6,7,8,9,10,11,12,13,14,15);
#pragma unroll
      for (int j = 0; j < 8; ++j) {
        // B frag: column n = j*16+l15, k = kc + 16*half .. +15 contiguous
        const bf16* bsrc = Bs + (j * 16 + l15) * BSTRIDE + kc + 16 * half;
        v16bf b = *(const v16bf*)bsrc;
        acc[j] = __builtin_amdgcn_wmma_f32_16x16x32_bf16(false, a, false, b,
                                                         (short)0, acc[j], false, false);
      }
    }
    __syncthreads();                          // chunk consumed; safe to overwrite
  }

#pragma unroll
  for (int j = 0; j < 8; ++j) {
    const int nn = nbase + j * 16 + l15;
    const float bias = lb[nn];
#pragma unroll
    for (int r = 0; r < 8; ++r) {
      const int m = m0 + r + 8 * half;
      out[(size_t)m * VOCAB + nn] = acc[j][r] + bias;
    }
  }
}

// ---------------------------------------------------------------------------
extern "C" void kernel_launch(void* const* d_in, const int* in_sizes, int n_in,
                              void* d_out, int out_size, void* d_ws, size_t ws_size,
                              hipStream_t stream) {
  const int*   X   = (const int*)  d_in[0];
  const float* emb = (const float*)d_in[1];
  const float* Wx  = (const float*)d_in[2];
  const float* Wh  = (const float*)d_in[3];
  const float* lw  = (const float*)d_in[4];
  const float* lb  = (const float*)d_in[5];
  float* out = (float*)d_out;

  // workspace layout
  const size_t OFF_XW  = 256;                                   // sync in [0,256)
  const size_t SZ_XW   = (size_t)TT * BB * HID * 4;             // 33,554,432
  const size_t OFF_H   = OFF_XW + SZ_XW;
  const size_t SZ_H    = (size_t)2 * BB * HID * 4;              // 131,072
  const size_t OFF_HSB = OFF_H + SZ_H;
  const size_t SZ_HSB  = (size_t)BB * TT * HID * 2;             // 16,777,216
  const size_t OFF_LWB = OFF_HSB + SZ_HSB;
  const size_t SZ_LWB  = (size_t)VOCAB * HID * 2;               // 65,536,000
  if (ws_size < OFF_LWB + SZ_LWB) return;                       // ws too small

  char*  ws   = (char*)d_ws;
  int*   sync = (int*)ws;
  float* xW   = (float*)(ws + OFF_XW);
  float* hbuf = (float*)(ws + OFF_H);
  bf16*  hsb  = (bf16*)(ws + OFF_HSB);
  bf16*  lwb  = (bf16*)(ws + OFF_LWB);

  k_convert_w<<<1024, 256, 0, stream>>>(lw, lwb, VOCAB * HID, sync);
  k_embed_xw <<<1024, 256, 0, stream>>>(X, emb, Wx, xW);
  k_rnn      <<<8,    256, 0, stream>>>(xW, Wh, hbuf, hsb, sync);
  k_logits   <<<dim3(250, 64), 256, 0, stream>>>(hsb, lwb, lb, out);
}
